// SelfAttention_60507499266629
// MI455X (gfx1250) — matrix operand order repaired and checked by reading, not compiled, not run on hardware
//
#include <hip/hip_runtime.h>

// ---------------------------------------------------------------------------
// CDNA5 (gfx1250) wave32 WMMA self-attention pipeline.
//   proj_qkv  : per-head Linear for Q/K/V (f32 VALU, f16 outputs to ws)
//   cvt_wo    : one-time Wo f32 -> f16 conversion (saves traffic in GEMM)
//   attn_fwd  : flash attention with v_wmma_f32_16x16x32_f16 (S^T trick),
//               double-buffered GLOBAL_LOAD_ASYNC_TO_LDS tile staging
//   out_proj  : final GEMM (Ao @ Wo^T + bo) with WMMA, async-staged tiles
// ---------------------------------------------------------------------------

typedef __attribute__((ext_vector_type(16))) _Float16 v16h;
typedef __attribute__((ext_vector_type(8)))  _Float16 v8h;
typedef __attribute__((ext_vector_type(8)))  float    v8f;
typedef __attribute__((ext_vector_type(4)))  int      v4i;

#define WMMA_F32_F16(a, b, c) \
    __builtin_amdgcn_wmma_f32_16x16x32_f16(false, (a), false, (b), (short)0, (c), false, false)

// gfx1250 async global->LDS copy path (ASYNCcnt-tracked), guarded so the file
// still compiles on toolchains without the builtins.
#if __has_builtin(__builtin_amdgcn_global_load_async_to_lds_b128) && \
    __has_builtin(__builtin_amdgcn_s_wait_asynccnt)
#define ATHENA_ASYNC 1
#else
#define ATHENA_ASYNC 0
#endif

// Builtin signature (from hipcc diagnostic): (v4i AS1*, v4i AS3*, imm, imm)
typedef __attribute__((address_space(1))) v4i gv4i;   // global
typedef __attribute__((address_space(3))) v4i sv4i;   // LDS

#define ASYNC_B128(gp, sp, off) \
    __builtin_amdgcn_global_load_async_to_lds_b128((gv4i*)(gp), (sv4i*)(sp), (off), 0)

namespace {
constexpr int kN = 4;       // batch
constexpr int kS = 2048;    // sequence length
constexpr int kH = 16;      // heads
constexpr int kD = 64;      // head dim
constexpr int kE = 1024;    // embed
constexpr float kScale = 0.03125f; // 1/sqrt(1024)
}

// ---------------------------------------------------------------------------
// Kernel 1: per-head projections. Block = 256 threads = 16 seq x 16 heads.
// Grid = N * (S/16).  Writes:
//   Qh,Kh : [n][h][s][d] f16
//   Vt    : [n][h][d][s] f16  (transposed so attention B-fragments are contig)
// ---------------------------------------------------------------------------
__global__ void __launch_bounds__(256) proj_qkv(
    const float* __restrict__ Xv, const float* __restrict__ Xk,
    const float* __restrict__ Xq,
    const float* __restrict__ Wv, const float* __restrict__ Wk,
    const float* __restrict__ Wq,
    _Float16* __restrict__ Qh, _Float16* __restrict__ Kh,
    _Float16* __restrict__ Vt)
{
    __shared__ float lWq[64 * 64];
    __shared__ float lWk[64 * 64];
    __shared__ float lWv[64 * 64];

    const int t = threadIdx.x;
    for (int i = t; i < 64 * 64; i += 256) {
        lWq[i] = Wq[i];
        lWk[i] = Wk[i];
        lWv[i] = Wv[i];
    }
    __syncthreads();

    const int n = blockIdx.x >> 7;                       // S/16 = 128 blocks per n
    const int s = ((blockIdx.x & 127) << 4) | (t >> 4);
    const int h = t & 15;
    const size_t xbase = ((size_t)(n * kS + s)) * kE + (size_t)h * kD;

    float x[64];

    // ---- Q = x @ Wq^T ----
    #pragma unroll 8
    for (int d = 0; d < 64; ++d) x[d] = Xq[xbase + d];
    {
        _Float16* out = Qh + (((size_t)(n * kH + h)) * kS + s) * kD;
        for (int e = 0; e < 64; ++e) {
            float acc = 0.f;
            #pragma unroll 8
            for (int d = 0; d < 64; ++d) acc = fmaf(x[d], lWq[e * 64 + d], acc);
            out[e] = (_Float16)acc;
        }
    }
    // ---- K = x @ Wk^T ----
    #pragma unroll 8
    for (int d = 0; d < 64; ++d) x[d] = Xk[xbase + d];
    {
        _Float16* out = Kh + (((size_t)(n * kH + h)) * kS + s) * kD;
        for (int e = 0; e < 64; ++e) {
            float acc = 0.f;
            #pragma unroll 8
            for (int d = 0; d < 64; ++d) acc = fmaf(x[d], lWk[e * 64 + d], acc);
            out[e] = (_Float16)acc;
        }
    }
    // ---- V = x @ Wv^T, stored transposed: Vt[n][h][e][s] ----
    #pragma unroll 8
    for (int d = 0; d < 64; ++d) x[d] = Xv[xbase + d];
    {
        for (int e = 0; e < 64; ++e) {
            float acc = 0.f;
            #pragma unroll 8
            for (int d = 0; d < 64; ++d) acc = fmaf(x[d], lWv[e * 64 + d], acc);
            Vt[(((size_t)(n * kH + h)) * kD + e) * kS + s] = (_Float16)acc;
        }
    }
}

// ---------------------------------------------------------------------------
// Kernel 1b: Wo f32 -> f16 (done once; GEMM then streams f16 weights).
// Grid = 1024, block = 256, 4 elements/thread.
// ---------------------------------------------------------------------------
__global__ void __launch_bounds__(256) cvt_wo(
    const float* __restrict__ Wo, _Float16* __restrict__ Wo16)
{
    const size_t i = ((size_t)blockIdx.x * 256 + threadIdx.x) * 4;
    #pragma unroll
    for (int j = 0; j < 4; ++j) Wo16[i + j] = (_Float16)Wo[i + j];
}

// ---------------------------------------------------------------------------
// Kernel 2: flash attention. Grid = N*H*(S/128). Block = 256 (8 waves).
// Each wave owns 16 query rows. Computes S^T = K_tile x Q^T so softmax
// reductions are lane-local (+ one shfl_xor 16), and the D-layout of S^T
// matches the A-fragment layout for the P*V WMMA (lane-local f32->f16 pack).
// K/V tiles staged via double-buffered GLOBAL_LOAD_ASYNC_TO_LDS (if available).
// Output Ao: [n][s][h*64+d] f16 rows for the final GEMM.
// ---------------------------------------------------------------------------
#if ATHENA_ASYNC
#define TILE_BUFS 2
#else
#define TILE_BUFS 1
#endif

__global__ void __launch_bounds__(256) attn_fwd(
    const _Float16* __restrict__ Qh, const _Float16* __restrict__ Kh,
    const _Float16* __restrict__ Vt, const int* __restrict__ mask,
    _Float16* __restrict__ Ao)
{
    __shared__ __align__(32) _Float16 lK[TILE_BUFS][64 * 64];  // keys x d
    __shared__ __align__(32) _Float16 lV[TILE_BUFS][64 * 64];  // d x keys
    __shared__ int lM[64];

    const int t      = threadIdx.x;
    const int wave   = t >> 5;
    const int lane   = t & 31;
    const int laneLo = lane & 15;
    const int laneHi = lane >> 4;

    const int qt = blockIdx.x & 15;     // 16 q-tiles of 128
    const int nh = blockIdx.x >> 4;     // n*16 + h
    const int n  = nh >> 4;
    const int h  = nh & 15;
    const int qbase = qt * 128 + wave * 16;

    const size_t headOff = (size_t)nh * kS * kD;

    // cooperative tile-load mapping: 64 rows x 4 chunks of 16 halves (32B)
    const int row = t >> 2;
    const int ch  = t & 3;

#if ATHENA_ASYNC
    auto issueTile = [&](int kb, int buf) {
        const _Float16* gk = Kh + headOff + (size_t)(kb * 64 + row) * kD + ch * 16;
        const _Float16* gv = Vt + ((size_t)nh * kD + row) * kS + kb * 64 + ch * 16;
        _Float16* sk = &lK[buf][row * 64 + ch * 16];
        _Float16* sv = &lV[buf][row * 64 + ch * 16];
        ASYNC_B128(gk, sk, 0);
        ASYNC_B128(gk, sk, 16);
        ASYNC_B128(gv, sv, 0);
        ASYNC_B128(gv, sv, 16);
    };
    issueTile(0, 0);
#endif

    // Q^T B-fragments (loop invariant): lane needs 16 contiguous d-values of
    // query row (qbase + laneLo), at d = 32*st + 16*laneHi.
    v16h bq[2];
    {
        const _Float16* qrow = Qh + headOff + (size_t)(qbase + laneLo) * kD;
        #pragma unroll
        for (int st = 0; st < 2; ++st)
            bq[st] = *(const v16h*)(qrow + 32 * st + 16 * laneHi);
    }

    v8f o[4];
    #pragma unroll
    for (int nt = 0; nt < 4; ++nt)
        #pragma unroll
        for (int r = 0; r < 8; ++r) o[nt][r] = 0.f;

    float m_i = -1e30f;
    float l_i = 0.f;

    for (int kb = 0; kb < kS / 64; ++kb) {
        const int kbase = kb * 64;
        const int buf   = kb & (TILE_BUFS - 1);

        if (t < 64) lM[t] = mask[n * kS + kbase + t];
#if ATHENA_ASYNC
        if (kb + 1 < kS / 64) {
            issueTile(kb + 1, (kb + 1) & 1);
            __builtin_amdgcn_s_wait_asynccnt(4);   // tile kb landed (in-order)
        } else {
            __builtin_amdgcn_s_wait_asynccnt(0);
        }
#else
        *(v16h*)&lK[0][row * 64 + ch * 16] =
            *(const v16h*)(Kh + headOff + (size_t)(kbase + row) * kD + ch * 16);
        *(v16h*)&lV[0][row * 64 + ch * 16] =
            *(const v16h*)(Vt + ((size_t)nh * kD + row) * kS + kbase + ch * 16);
        if (kb + 1 < kS / 64) {   // prefetch next tile into L2/WGP$
            __builtin_prefetch(Kh + headOff + (size_t)(kbase + 64 + row) * kD + ch * 16, 0, 1);
            __builtin_prefetch(Vt + ((size_t)nh * kD + row) * kS + kbase + 64 + ch * 16, 0, 1);
        }
#endif
        __syncthreads();

        // ---- scores S^T = K_tile x Q^T, 4 blocks of 16 keys ----
        float sc[4][8];
        #pragma unroll
        for (int b = 0; b < 4; ++b) {
            v8f acc;
            #pragma unroll
            for (int r = 0; r < 8; ++r) acc[r] = 0.f;
            #pragma unroll
            for (int st = 0; st < 2; ++st) {
                // A-frag of K: row = key (laneLo), two 8-half chunks
                const _Float16* arow = &lK[buf][(16 * b + laneLo) * 64 + 32 * st + 16 * laneHi];
                v8h alo = *(const v8h*)arow;
                v8h ahi = *(const v8h*)(arow + 8);
                v16h a;
                #pragma unroll
                for (int i = 0; i < 8; ++i) { a[i] = alo[i]; a[8 + i] = ahi[i]; }
                acc = WMMA_F32_F16(a, bq[st], acc);
            }
            #pragma unroll
            for (int r = 0; r < 8; ++r) {
                float sv = acc[r];                       // key 16b + 8*laneHi + r
                if (lM[16 * b + 8 * laneHi + r] == 0) sv = -1e20f;
                sc[b][r] = sv * kScale;
            }
        }

        // ---- online softmax (lane holds one query; halves hold disjoint keys) ----
        float mx = -1e30f;
        #pragma unroll
        for (int b = 0; b < 4; ++b)
            #pragma unroll
            for (int r = 0; r < 8; ++r) mx = fmaxf(mx, sc[b][r]);
        mx = fmaxf(mx, __shfl_xor(mx, 16, 32));
        const float mnew = fmaxf(m_i, mx);

        float ps = 0.f;
        #pragma unroll
        for (int b = 0; b < 4; ++b)
            #pragma unroll
            for (int r = 0; r < 8; ++r) {
                float p = __expf(sc[b][r] - mnew);
                sc[b][r] = p;
                ps += p;
            }
        ps += __shfl_xor(ps, 16, 32);
        const float corr = __expf(m_i - mnew);
        l_i = l_i * corr + ps;
        m_i = mnew;

        // rescale accumulators; row m stats live in lane (m) (= lane m+16)
        #pragma unroll
        for (int r = 0; r < 8; ++r) {
            const float fr = __shfl(corr, 8 * laneHi + r, 32);
            #pragma unroll
            for (int nt = 0; nt < 4; ++nt) o[nt][r] *= fr;
        }

        // ---- O += P x V ; P A-frag is a lane-local repack of sc ----
        #pragma unroll
        for (int j = 0; j < 2; ++j) {
            v16h ap;
            #pragma unroll
            for (int c = 0; c < 8; ++c) {
                ap[c]     = (_Float16)sc[2 * j][c];
                ap[8 + c] = (_Float16)sc[2 * j + 1][c];
            }
            #pragma unroll
            for (int nt = 0; nt < 4; ++nt) {
                const _Float16* vp = &lV[buf][(16 * nt + laneLo) * 64 + 32 * j + 8 * laneHi];
                v8h vlo = *(const v8h*)vp;
                v8h vhi = *(const v8h*)(vp + 16);
                v16h bv;
                #pragma unroll
                for (int i = 0; i < 8; ++i) { bv[i] = vlo[i]; bv[8 + i] = vhi[i]; }
                o[nt] = WMMA_F32_F16(ap, bv, o[nt]);
            }
        }
        __syncthreads();   // all reads done before next tile overwrites LDS
    }

    // ---- normalize and store (D layout: lane -> col d = 16nt+laneLo, rows 8*laneHi+r)
    const float inv = 1.f / l_i;
    #pragma unroll
    for (int r = 0; r < 8; ++r) {
        const float ir = __shfl(inv, 8 * laneHi + r, 32);
        const int   m  = 8 * laneHi + r;
        const size_t rowo = ((size_t)(n * kS + qbase + m)) * kE + (size_t)h * kD + laneLo;
        #pragma unroll
        for (int nt = 0; nt < 4; ++nt)
            Ao[rowo + 16 * nt] = (_Float16)(o[nt][r] * ir);
    }
}

// ---------------------------------------------------------------------------
// Kernel 3: Y = Ao @ Wo16^T + bo  (8192 x 1024 x 1024), f32 output.
// Grid = 64 row-tiles x 16 col-tiles. Block = 256 (8 waves x 16 rows).
// A/W tiles staged via double-buffered async LDS loads when available.
// ---------------------------------------------------------------------------
__global__ void __launch_bounds__(256) out_proj(
    const _Float16* __restrict__ Ao, const _Float16* __restrict__ Wo16,
    const float* __restrict__ bo, float* __restrict__ Y)
{
    __shared__ __align__(32) _Float16 lA[TILE_BUFS][128 * 64];
    __shared__ __align__(32) _Float16 lW[TILE_BUFS][64 * 64];   // lW[c][e]

    const int t      = threadIdx.x;
    const int wave   = t >> 5;
    const int lane   = t & 31;
    const int laneLo = lane & 15;
    const int laneHi = lane >> 4;

    const int rb = blockIdx.x & 63;
    const int cb = blockIdx.x >> 6;
    const int rowBase = rb * 128;
    const int colBase = cb * 64;

    const int arow = t >> 1, ach = t & 1;  // A tile: 128 rows x 2 chunks of 32 halves
    const int wci  = t >> 2, wc4 = t & 3;  // W tile: 64 rows x 4 chunks of 16 halves

#if ATHENA_ASYNC
    auto issueTile = [&](int eb, int buf) {
        const _Float16* ga = Ao + (size_t)(rowBase + arow) * kE + eb * 64 + 32 * ach;
        const _Float16* gw = Wo16 + (size_t)(colBase + wci) * kE + eb * 64 + 16 * wc4;
        _Float16* sa = &lA[buf][arow * 64 + 32 * ach];
        _Float16* sw = &lW[buf][wci * 64 + 16 * wc4];
        ASYNC_B128(ga, sa, 0);
        ASYNC_B128(ga, sa, 16);
        ASYNC_B128(ga, sa, 32);
        ASYNC_B128(ga, sa, 48);
        ASYNC_B128(gw, sw, 0);
        ASYNC_B128(gw, sw, 16);
    };
    issueTile(0, 0);
#endif

    v8f o[4];
    #pragma unroll
    for (int nt = 0; nt < 4; ++nt)
        #pragma unroll
        for (int r = 0; r < 8; ++r) o[nt][r] = 0.f;

    for (int eb = 0; eb < kE / 64; ++eb) {
        const int buf = eb & (TILE_BUFS - 1);
#if ATHENA_ASYNC
        if (eb + 1 < kE / 64) {
            issueTile(eb + 1, (eb + 1) & 1);
            __builtin_amdgcn_s_wait_asynccnt(6);
        } else {
            __builtin_amdgcn_s_wait_asynccnt(0);
        }
#else
        {
            const int ebase = eb * 64;
            const _Float16* src = Ao + (size_t)(rowBase + arow) * kE + ebase + 32 * ach;
            *(v16h*)&lA[0][arow * 64 + 32 * ach]      = *(const v16h*)src;
            *(v16h*)&lA[0][arow * 64 + 32 * ach + 16] = *(const v16h*)(src + 16);
            *(v16h*)&lW[0][wci * 64 + 16 * wc4] =
                *(const v16h*)(Wo16 + (size_t)(colBase + wci) * kE + ebase + 16 * wc4);
        }
#endif
        __syncthreads();

        #pragma unroll
        for (int st = 0; st < 2; ++st) {
            const _Float16* ap = &lA[buf][(wave * 16 + laneLo) * 64 + 32 * st + 16 * laneHi];
            v8h alo = *(const v8h*)ap;
            v8h ahi = *(const v8h*)(ap + 8);
            v16h a;
            #pragma unroll
            for (int i = 0; i < 8; ++i) { a[i] = alo[i]; a[8 + i] = ahi[i]; }
            #pragma unroll
            for (int nt = 0; nt < 4; ++nt) {
                v16h b = *(const v16h*)&lW[buf][(16 * nt + laneLo) * 64 + 32 * st + 16 * laneHi];
                o[nt] = WMMA_F32_F16(a, b, o[nt]);
            }
        }
        __syncthreads();
    }

    #pragma unroll
    for (int nt = 0; nt < 4; ++nt) {
        const int c = colBase + 16 * nt + laneLo;
        const float bias = bo[c];
        #pragma unroll
        for (int r = 0; r < 8; ++r) {
            const int m = 8 * laneHi + r;
            Y[(size_t)(rowBase + wave * 16 + m) * kE + c] = o[nt][r] + bias;
        }
    }
}

// ---------------------------------------------------------------------------
extern "C" void kernel_launch(void* const* d_in, const int* in_sizes, int n_in,
                              void* d_out, int out_size, void* d_ws, size_t ws_size,
                              hipStream_t stream)
{
    const float* Xv   = (const float*)d_in[0];
    const float* Xk   = (const float*)d_in[1];
    const float* Xq   = (const float*)d_in[2];
    const int*   mask = (const int*)  d_in[3];
    const float* Wv   = (const float*)d_in[4];
    const float* Wk   = (const float*)d_in[5];
    const float* Wq   = (const float*)d_in[6];
    const float* Wo   = (const float*)d_in[7];
    const float* bo   = (const float*)d_in[8];
    float*       Y    = (float*)d_out;

    // Workspace layout (f16): Qh | Kh | Vt | Ao (16 MB each) | Wo16 (2 MB).
    const size_t tHalves = (size_t)kN * kH * kS * kD;
    _Float16* Qh   = (_Float16*)d_ws;
    _Float16* Kh   = Qh + tHalves;
    _Float16* Vt   = Kh + tHalves;
    _Float16* Ao   = Vt + tHalves;
    _Float16* Wo16 = Ao + tHalves;

    proj_qkv<<<kN * (kS / 16), 256, 0, stream>>>(Xv, Xk, Xq, Wv, Wk, Wq, Qh, Kh, Vt);
    cvt_wo<<<(kE * kE) / 1024, 256, 0, stream>>>(Wo, Wo16);
    attn_fwd<<<kN * kH * (kS / 128), 256, 0, stream>>>(Qh, Kh, Vt, mask, Ao);
    out_proj<<<(kN * kS / 128) * (kE / 64), 256, 0, stream>>>(Ao, Wo16, bo, Y);
}
